// MultiHeadAttention_29592324670265
// MI455X (gfx1250) — compile-verified, matrix operand order checked by reading, not run on hardware
//
#include <hip/hip_runtime.h>
#include <hip/hip_bf16.h>

// ---------- types ----------
typedef __attribute__((ext_vector_type(16))) __bf16 v16bf;
typedef __attribute__((ext_vector_type(8)))  __bf16 v8bf;
typedef __attribute__((ext_vector_type(4)))  __bf16 v4bf;
typedef __attribute__((ext_vector_type(8)))  float  v8f;
typedef __attribute__((ext_vector_type(4)))  int    v4i;

#define D_MODEL 1024
#define NUM_HEADS 16
#define HEAD_DIM 64
#define BATCH 4
#define SEQ 4096
#define TOKENS (BATCH * SEQ)

// GEMM tiling
#define BM 128
#define BN 128
#define KT 64
#define STR 72   // LDS row stride in bf16 (144B) - conflict padding, 16B aligned

// ---------- gfx1250 async global->LDS path (guarded) ----------
#if __has_builtin(__builtin_amdgcn_global_load_async_to_lds_b128)
#define HAVE_ASYNC_LDS 1
#endif

#if __has_builtin(__builtin_amdgcn_s_wait_asynccnt)
#define WAIT_ASYNC() __builtin_amdgcn_s_wait_asynccnt(0)
#else
#define WAIT_ASYNC() asm volatile("s_wait_asynccnt 0x0" ::: "memory")
#endif

#ifdef HAVE_ASYNC_LDS
// builtin signature (from probe diagnostics): (v4i AS1*, v4i AS3*, imm offset, imm cpol)
typedef __attribute__((address_space(1))) v4i* gbl_v4i_p;
typedef __attribute__((address_space(3))) v4i* lds_v4i_p;

__device__ __forceinline__ void async_copy_b128(const __bf16* g, __bf16* l) {
  // global generic ptr == AS1 numeric value; LDS offset == low 32 bits of
  // generic LDS-aperture address
  gbl_v4i_p gp = (gbl_v4i_p)(unsigned long long)g;
  lds_v4i_p lp = (lds_v4i_p)(unsigned)(unsigned long long)l;
  __builtin_amdgcn_global_load_async_to_lds_b128(gp, lp, 0, 0);
}
#endif

// ---------- helpers ----------
__device__ __forceinline__ __bf16 f2bf(float f) {
  union { float f; unsigned u; } a; a.f = f;
  unsigned u = a.u;
  unsigned short h = (unsigned short)((u + 0x7FFFu + ((u >> 16) & 1u)) >> 16);
  union { unsigned short s; __bf16 b; } o; o.s = h;
  return o.b;
}

__device__ __forceinline__ v16bf make_frag(v8bf lo, v8bf hi) {
  v16bf r;
#pragma unroll
  for (int i = 0; i < 8; i++) { r[i] = lo[i]; r[i + 8] = hi[i]; }
  return r;
}

__device__ __forceinline__ void store_elem(float* C, size_t idx, float v)  { C[idx] = v; }
__device__ __forceinline__ void store_elem(__bf16* C, size_t idx, float v) { C[idx] = f2bf(v); }

// ---------- tile staging (software pipeline) ----------
// f32 source: register-staged (global -> VGPR early, cvt+ds_store late)
// bf16 source: async DMA to LDS if available, else register-staged
template <typename T> struct StageT;
template <> struct StageT<float>  { float4 r[8]; };   // 128x64 f32 / 256 thr = 32 f32
template <> struct StageT<__bf16> { v8bf  r[4]; };    // 128x64 bf16 / 256 thr = 32 bf16

__device__ __forceinline__ void stage_issue(const float* __restrict__ src, int K, int kt,
                                            int tid, StageT<float>& st, __bf16* lds) {
  (void)lds;
#pragma unroll
  for (int i = 0; i < 8; i++) {
    int c = tid + i * 256;                 // 2048 float4 chunks
    int row = c >> 4, col4 = (c & 15) * 4;
    st.r[i] = *reinterpret_cast<const float4*>(&src[(size_t)row * K + kt + col4]);
  }
}
__device__ __forceinline__ void stage_commit(int tid, const StageT<float>& st, __bf16* lds) {
#pragma unroll
  for (int i = 0; i < 8; i++) {
    int c = tid + i * 256;
    int row = c >> 4, col4 = (c & 15) * 4;
    __bf16* d = &lds[row * STR + col4];
    d[0] = f2bf(st.r[i].x); d[1] = f2bf(st.r[i].y);
    d[2] = f2bf(st.r[i].z); d[3] = f2bf(st.r[i].w);
  }
}

__device__ __forceinline__ void stage_issue(const __bf16* __restrict__ src, int K, int kt,
                                            int tid, StageT<__bf16>& st, __bf16* lds) {
#pragma unroll
  for (int i = 0; i < 4; i++) {
    int c = tid + i * 256;                 // 1024 b128 chunks
    int row = c >> 3, cc = (c & 7) * 8;
#ifdef HAVE_ASYNC_LDS
    (void)st;
    async_copy_b128(&src[(size_t)row * K + kt + cc], &lds[row * STR + cc]);
#else
    (void)lds;
    st.r[i] = *reinterpret_cast<const v8bf*>(&src[(size_t)row * K + kt + cc]);
#endif
  }
}
__device__ __forceinline__ void stage_commit(int tid, const StageT<__bf16>& st, __bf16* lds) {
#ifdef HAVE_ASYNC_LDS
  (void)tid; (void)st; (void)lds;
#else
#pragma unroll
  for (int i = 0; i < 4; i++) {
    int c = tid + i * 256;
    int row = c >> 3, cc = (c & 7) * 8;
    *reinterpret_cast<v8bf*>(&lds[row * STR + cc]) = st.r[i];
  }
#endif
}

// ---------- f32 -> bf16 conversion (weights) ----------
__global__ __launch_bounds__(256)
void cvt_f32_bf16_kernel(const float* __restrict__ src, __bf16* __restrict__ dst) {
  size_t i = ((size_t)blockIdx.x * 256 + threadIdx.x) * 4;
  float4 v = *reinterpret_cast<const float4*>(&src[i]);
  __bf16* d = &dst[i];
  d[0] = f2bf(v.x); d[1] = f2bf(v.y); d[2] = f2bf(v.z); d[3] = f2bf(v.w);
}

// ---------- GEMM: C[M,N] = X[M,K] @ Wb[N,K]^T + bias ----------
// block 128x128, 8 waves of 32(M)x64(N), KT=64, double-buffered LDS pipeline
template <typename TIN, typename TOUT>
__global__ __launch_bounds__(256)
void gemm_bias_kernel(const TIN* __restrict__ X, const __bf16* __restrict__ Wb,
                      const float* __restrict__ bias, TOUT* __restrict__ C,
                      int M, int N, int K) {
  __shared__ __bf16 As[2][BM * STR];
  __shared__ __bf16 Bs[2][BN * STR];

  const int tid  = threadIdx.x;
  const int lane = tid & 31;
  const int wave = tid >> 5;
  const int wm = wave & 3;
  const int wn = wave >> 2;
  const int m0 = blockIdx.y * BM;
  const int n0 = blockIdx.x * BN;

  const int lm    = lane & 15;
  const int halft = lane >> 4;
  const int k0    = halft * 8;

  const TIN*    Xblk = X  + (size_t)m0 * K;
  const __bf16* Wblk = Wb + (size_t)n0 * K;

  v8f acc[2][4];
#pragma unroll
  for (int i = 0; i < 2; i++)
#pragma unroll
    for (int j = 0; j < 4; j++) acc[i][j] = v8f{};

  StageT<TIN>    ast;
  StageT<__bf16> wst;

  // prologue: stage 0 into buffer 0
  stage_issue(Xblk, K, 0, tid, ast, As[0]);
  stage_issue(Wblk, K, 0, tid, wst, Bs[0]);
  stage_commit(tid, ast, As[0]);
  stage_commit(tid, wst, Bs[0]);
  WAIT_ASYNC();
  __syncthreads();

  int buf = 0;
  for (int kt = 0; kt < K; kt += KT) {
    const bool next = (kt + KT < K);
    if (next) {
      stage_issue(Xblk, K, kt + KT, tid, ast, As[buf ^ 1]);  // overlaps compute
      stage_issue(Wblk, K, kt + KT, tid, wst, Bs[buf ^ 1]);
    }

    const __bf16* as = As[buf];
    const __bf16* bs = Bs[buf];
#pragma unroll
    for (int ks = 0; ks < 2; ks++) {
      const int kb = ks * 32 + k0;
      v16bf afr[2], bfr[4];
#pragma unroll
      for (int mt = 0; mt < 2; mt++) {
        int m = wm * 32 + mt * 16 + lm;
        afr[mt] = make_frag(*(const v8bf*)&as[m * STR + kb],
                            *(const v8bf*)&as[m * STR + kb + 16]);
      }
#pragma unroll
      for (int nt = 0; nt < 4; nt++) {
        int n = wn * 64 + nt * 16 + lm;
        bfr[nt] = make_frag(*(const v8bf*)&bs[n * STR + kb],
                            *(const v8bf*)&bs[n * STR + kb + 16]);
      }
#pragma unroll
      for (int mt = 0; mt < 2; mt++)
#pragma unroll
        for (int nt = 0; nt < 4; nt++)
          acc[mt][nt] = __builtin_amdgcn_wmma_f32_16x16x32_bf16(
              false, afr[mt], false, bfr[nt], (short)0, acc[mt][nt], false, false);
    }

    if (next) {
      stage_commit(tid, ast, As[buf ^ 1]);
      stage_commit(tid, wst, Bs[buf ^ 1]);
      WAIT_ASYNC();
    }
    __syncthreads();
    buf ^= 1;
  }

  // epilogue: C layout (ISA 7.12.2): col = lane&15, row = r + 8*(lane>>4)
#pragma unroll
  for (int nt = 0; nt < 4; nt++) {
    int n = n0 + wn * 64 + nt * 16 + lm;
    float bv = bias[n];
#pragma unroll
    for (int mt = 0; mt < 2; mt++) {
#pragma unroll
      for (int r = 0; r < 8; r++) {
        int m = m0 + wm * 32 + mt * 16 + r + halft * 8;
        store_elem(C, (size_t)m * N + n, acc[mt][nt][r] + bv);
      }
    }
  }
}

// ---------- per-token attention: E = Qt Kt^T, softmax rows, Y = Att Vt ----------
__global__ __launch_bounds__(256)
void attn_kernel(const __bf16* __restrict__ Q, const __bf16* __restrict__ K,
                 const __bf16* __restrict__ V, const int* __restrict__ mask,
                 __bf16* __restrict__ O) {
  __shared__ float attLds[8][16][16];
  const int tid   = threadIdx.x;
  const int lane  = tid & 31;
  const int wave  = tid >> 5;
  const int tok   = blockIdx.x * 8 + wave;
  const size_t base = (size_t)tok * D_MODEL;
  const int lm    = lane & 15;
  const int halft = lane >> 4;
  const int k0    = halft * 8;

  v8f e = v8f{};
#pragma unroll
  for (int kk = 0; kk < 2; kk++) {
    v16bf aq = make_frag(*(const v8bf*)&Q[base + (size_t)lm * 64 + kk * 32 + k0],
                         *(const v8bf*)&Q[base + (size_t)lm * 64 + kk * 32 + k0 + 16]);
    v16bf bk = make_frag(*(const v8bf*)&K[base + (size_t)lm * 64 + kk * 32 + k0],
                         *(const v8bf*)&K[base + (size_t)lm * 64 + kk * 32 + k0 + 16]);
    e = __builtin_amdgcn_wmma_f32_16x16x32_bf16(false, aq, false, bk, (short)0, e,
                                                false, false);
  }

  const int mv = mask[tok];   // wave-uniform
  float att[8];
  if (mv == 0) {
#pragma unroll
    for (int r = 0; r < 8; r++) att[r] = 1.0f / 16.0f;
  } else {
#pragma unroll
    for (int r = 0; r < 8; r++) {
      float x = e[r] * 0.03125f;  // 1/sqrt(1024)
      float mx = x;
#pragma unroll
      for (int sh = 1; sh < 16; sh <<= 1) mx = fmaxf(mx, __shfl_xor(mx, sh, 32));
      float p = __expf(x - mx);
      float s = p;
#pragma unroll
      for (int sh = 1; sh < 16; sh <<= 1) s += __shfl_xor(s, sh, 32);
      att[r] = p / s;
    }
  }

#pragma unroll
  for (int r = 0; r < 8; r++) attLds[wave][r + halft * 8][lm] = att[r];
  __syncthreads();

  v16bf a2;
#pragma unroll
  for (int t = 0; t < 8; t++) {
    a2[t]     = f2bf(attLds[wave][lm][k0 + t]);
    a2[t + 8] = f2bf(0.f);
  }

#pragma unroll
  for (int c = 0; c < 4; c++) {
    v16bf b2;
#pragma unroll
    for (int t = 0; t < 8; t++) {
      b2[t]     = V[base + (size_t)(k0 + t) * 64 + c * 16 + lm];
      b2[t + 8] = f2bf(0.f);
    }
    v8f zero = v8f{};
    v8f y = __builtin_amdgcn_wmma_f32_16x16x32_bf16(false, a2, false, b2, (short)0,
                                                    zero, false, false);
#pragma unroll
    for (int r = 0; r < 8; r++) {
      int q = r + halft * 8;
      O[base + (size_t)q * 64 + c * 16 + lm] = f2bf(y[r]);
    }
  }
}

// ---------- driver ----------
extern "C" void kernel_launch(void* const* d_in, const int* in_sizes, int n_in,
                              void* d_out, int out_size, void* d_ws, size_t ws_size,
                              hipStream_t stream) {
  (void)in_sizes; (void)n_in; (void)out_size; (void)ws_size;
  const float* query = (const float*)d_in[0];
  const float* key   = (const float*)d_in[1];
  const float* value = (const float*)d_in[2];
  const int*   mask  = (const int*)d_in[3];
  const float* Wq = (const float*)d_in[4];
  const float* bq = (const float*)d_in[5];
  const float* Wk = (const float*)d_in[6];
  const float* bk = (const float*)d_in[7];
  const float* Wv = (const float*)d_in[8];
  const float* bv = (const float*)d_in[9];
  const float* Wo = (const float*)d_in[10];
  const float* bo = (const float*)d_in[11];
  float* out = (float*)d_out;

  const int M = TOKENS, N = D_MODEL, Kd = D_MODEL;
  const size_t WSZ = (size_t)D_MODEL * D_MODEL;   // 1M elements per weight
  const size_t ASZ = (size_t)M * N;               // 16.7M elements per activation

  __bf16* Wqb = (__bf16*)d_ws;
  __bf16* Wkb = Wqb + WSZ;
  __bf16* Wvb = Wkb + WSZ;
  __bf16* Wob = Wvb + WSZ;
  __bf16* Qb  = Wob + WSZ;
  __bf16* Kb  = Qb + ASZ;
  __bf16* Vb  = Kb + ASZ;
  __bf16* Ob  = Vb + ASZ;                         // total ~142 MB workspace

  // weights -> bf16 once (stay L2-resident: 8 MB total vs 192 MB L2)
  const int cvtBlocks = (int)(WSZ / (4 * 256));
  cvt_f32_bf16_kernel<<<cvtBlocks, 256, 0, stream>>>(Wq, Wqb);
  cvt_f32_bf16_kernel<<<cvtBlocks, 256, 0, stream>>>(Wk, Wkb);
  cvt_f32_bf16_kernel<<<cvtBlocks, 256, 0, stream>>>(Wv, Wvb);
  cvt_f32_bf16_kernel<<<cvtBlocks, 256, 0, stream>>>(Wo, Wob);

  dim3 gg(N / BN, M / BM);
  gemm_bias_kernel<float, __bf16><<<gg, 256, 0, stream>>>(query, Wqb, bq, Qb, M, N, Kd);
  gemm_bias_kernel<float, __bf16><<<gg, 256, 0, stream>>>(key,   Wkb, bk, Kb, M, N, Kd);
  gemm_bias_kernel<float, __bf16><<<gg, 256, 0, stream>>>(value, Wvb, bv, Vb, M, N, Kd);
  attn_kernel<<<TOKENS / 8, 256, 0, stream>>>(Qb, Kb, Vb, mask, Ob);
  gemm_bias_kernel<__bf16, float><<<gg, 256, 0, stream>>>(Ob, Wob, bo, out, M, N, Kd);
}
